// NoSoftmaxAttention_69896297775615
// MI455X (gfx1250) — matrix operand order repaired and checked by reading, compiled, not run on hardware
//
#include <hip/hip_runtime.h>

#define B_ 4
#define S_ 4096
#define E_ 1024
#define H_ 16
#define D_ 64
#define SCALE_ 0.125f   // 64^-0.5
#define NSPLIT_ 8       // S-dim split for kv reduction

typedef __attribute__((ext_vector_type(16))) __bf16 v16bf;
typedef __attribute__((ext_vector_type(8)))  float  v8f;

union Frag { v16bf v; unsigned int u[8]; uint4 q[2]; };

__device__ __forceinline__ unsigned short f2bf(float f) {
  unsigned int x = __builtin_bit_cast(unsigned int, f);
  x += 0x7FFFu + ((x >> 16) & 1u);          // round-to-nearest-even
  return (unsigned short)(x >> 16);
}

// ---------------------------------------------------------------------------
// One-time f32 -> bf16 conversion (4 elements / thread, vectorized)
// ---------------------------------------------------------------------------
__global__ __launch_bounds__(256) void cvt_f32_bf16(
    const float* __restrict__ s, unsigned short* __restrict__ d, int n4)
{
  const int i = blockIdx.x * 256 + threadIdx.x;
  if (i >= n4) return;
  const float4 f = ((const float4*)s)[i];
  uint2 o;
  o.x = (unsigned int)f2bf(f.x) | ((unsigned int)f2bf(f.y) << 16);
  o.y = (unsigned int)f2bf(f.z) | ((unsigned int)f2bf(f.w) << 16);
  ((uint2*)d)[i] = o;
}

// ---------------------------------------------------------------------------
// Tiled bf16 GEMM: C[M,N] = A[M,K] @ W[N,K]^T + bias    (torch Linear layout)
// 128x128 tile, 16 waves, each wave 2x2 WMMA tiles, K-chunks of 64 with
// register prefetch of the next chunk (global loads overlap WMMA compute).
// LDS rows padded to 72 halves (144B): 16B-aligned rows -> all fragment
// gathers are ds_load_b128; bank stride 36 is conflict-free over 16 rows.
// ---------------------------------------------------------------------------
template <bool OUT_BF16>
__global__ __launch_bounds__(512) void gemm_tn(
    const unsigned short* __restrict__ A, const unsigned short* __restrict__ W,
    const float* __restrict__ bias, void* __restrict__ Cptr,
    int M, int N, int K)
{
  __shared__ unsigned short a_lds[128][72];
  __shared__ unsigned short b_lds[128][72];

  const int tid  = threadIdx.x;
  const int lane = tid & 31;
  const int wave = tid >> 5;        // 0..15
  const int wr   = wave >> 2;       // wave row (M)
  const int wc   = wave & 3;        // wave col (N)
  const int m_base = blockIdx.y * 128;
  const int n_base = blockIdx.x * 128;

  const int lrow = tid >> 2;            // 0..127
  const int lcol = (tid & 3) << 4;      // 0,16,32,48 (halves)

  const int lh  = lane >> 4;            // lane half
  const int r16 = lane & 15;

  const unsigned short* Ap = A + (size_t)(m_base + lrow) * K + lcol;
  const unsigned short* Wp = W + (size_t)(n_base + lrow) * K + lcol;

  v8f acc[2][2] = {};

  // prime the pipeline: chunk 0 into registers
  uint4 pa0 = *(const uint4*)(Ap);
  uint4 pa1 = *(const uint4*)(Ap + 8);
  uint4 pb0 = *(const uint4*)(Wp);
  uint4 pb1 = *(const uint4*)(Wp + 8);

  for (int k0 = 0; k0 < K; k0 += 64) {
    // commit staged chunk to LDS
    *(uint4*)&a_lds[lrow][lcol]     = pa0;
    *(uint4*)&a_lds[lrow][lcol + 8] = pa1;
    *(uint4*)&b_lds[lrow][lcol]     = pb0;
    *(uint4*)&b_lds[lrow][lcol + 8] = pb1;
    __syncthreads();

    // issue next-chunk global loads; waited on at next LDS commit
    if (k0 + 64 < K) {
      pa0 = *(const uint4*)(Ap + k0 + 64);
      pa1 = *(const uint4*)(Ap + k0 + 72);
      pb0 = *(const uint4*)(Wp + k0 + 64);
      pb1 = *(const uint4*)(Wp + k0 + 72);
    }

#pragma unroll
    for (int c = 0; c < 2; ++c) {       // two 32-wide K sub-chunks
      Frag af[2], bfr[2];
#pragma unroll
      for (int i = 0; i < 2; ++i) {
        const unsigned short* r = a_lds[wr * 32 + i * 16 + r16] + c * 32;
        af[i].q[0] = *(const uint4*)(r + 8 * lh);        // K = 8lh..8lh+7
        af[i].q[1] = *(const uint4*)(r + 16 + 8 * lh);   // K = 16+8lh..
      }
#pragma unroll
      for (int j = 0; j < 2; ++j) {
        const unsigned short* r = b_lds[wc * 32 + j * 16 + r16] + c * 32;
        bfr[j].q[0] = *(const uint4*)(r + 8 * lh);       // K = 8lh..8lh+7
        bfr[j].q[1] = *(const uint4*)(r + 16 + 8 * lh);  // K = 16+8lh..
      }
#pragma unroll
      for (int i = 0; i < 2; ++i)
#pragma unroll
        for (int j = 0; j < 2; ++j)
          acc[i][j] = __builtin_amdgcn_wmma_f32_16x16x32_bf16(
              false, af[i].v, false, bfr[j].v, (short)0, acc[i][j], false, false);
    }
    __syncthreads();
  }

  // Epilogue: C layout — VGPR r holds m = r + 8*(lane/16), n = lane&15
#pragma unroll
  for (int j = 0; j < 2; ++j) {
    const int n  = n_base + wc * 32 + j * 16 + r16;
    const float bv = bias[n];
#pragma unroll
    for (int i = 0; i < 2; ++i) {
      const int m0 = m_base + wr * 32 + i * 16 + lh * 8;
#pragma unroll
      for (int r = 0; r < 8; ++r) {
        const float val = acc[i][j][r] + bv;
        if (OUT_BF16)
          ((unsigned short*)Cptr)[(size_t)(m0 + r) * N + n] = f2bf(val);
        else
          ((float*)Cptr)[(size_t)(m0 + r) * N + n] = val;
      }
    }
  }
}

// ---------------------------------------------------------------------------
// kv partials: KVpart[sp,b,h,d,e] = SCALE * sum_{s in chunk sp} K[b,s,h,d]*V[b,s,h,e]
// Grid (B*H, NSPLIT). K and V chunks staged TRANSPOSED into LDS so both
// operand fragments are aligned b128 loads.
// ---------------------------------------------------------------------------
__global__ __launch_bounds__(512) void kv_kernel(
    const unsigned short* __restrict__ Kb, const unsigned short* __restrict__ Vb,
    float* __restrict__ KVpart)
{
  __shared__ unsigned short kT[64][40];   // [d][s-chunk]
  __shared__ unsigned short vT[64][40];   // [e][s-chunk]

  const int bh = blockIdx.x;
  const int sp = blockIdx.y;
  const int b = bh / H_, h = bh % H_;
  const int tid  = threadIdx.x;
  const int lane = tid & 31;
  const int wave = tid >> 5;           // 0..15
  const int dt = (wave >> 2) * 16;
  const int et = (wave & 3) * 16;
  const int srow  = tid >> 4;          // 0..31 (s within chunk)
  const int scol4 = (tid & 15) << 2;   // 0..60 (d / e)
  const int lh = lane >> 4, r16 = lane & 15;

  v8f acc = {};
  const size_t base = (size_t)b * S_ * E_ + (size_t)h * D_;
  const int s_begin = sp * (S_ / NSPLIT_);
  const int s_end   = s_begin + (S_ / NSPLIT_);

  for (int s0 = s_begin; s0 < s_end; s0 += 32) {
    const uint2 ku = *(const uint2*)(Kb + base + (size_t)(s0 + srow) * E_ + scol4);
    const uint2 vu = *(const uint2*)(Vb + base + (size_t)(s0 + srow) * E_ + scol4);
    kT[scol4 + 0][srow] = (unsigned short)ku.x;
    kT[scol4 + 1][srow] = (unsigned short)(ku.x >> 16);
    kT[scol4 + 2][srow] = (unsigned short)ku.y;
    kT[scol4 + 3][srow] = (unsigned short)(ku.y >> 16);
    vT[scol4 + 0][srow] = (unsigned short)vu.x;
    vT[scol4 + 1][srow] = (unsigned short)(vu.x >> 16);
    vT[scol4 + 2][srow] = (unsigned short)vu.y;
    vT[scol4 + 3][srow] = (unsigned short)(vu.y >> 16);
    __syncthreads();

    Frag af, bfr;
    af.q[0]  = *(const uint4*)(kT[dt + r16] + 8 * lh);
    af.q[1]  = *(const uint4*)(kT[dt + r16] + 16 + 8 * lh);
    bfr.q[0] = *(const uint4*)(vT[et + r16] + 8 * lh);
    bfr.q[1] = *(const uint4*)(vT[et + r16] + 16 + 8 * lh);
    acc = __builtin_amdgcn_wmma_f32_16x16x32_bf16(
        false, af.v, false, bfr.v, (short)0, acc, false, false);
    __syncthreads();
  }

  float* out = KVpart + (size_t)sp * (B_ * H_ * D_ * D_) + (size_t)bh * D_ * D_;
#pragma unroll
  for (int r = 0; r < 8; ++r)
    out[(dt + lh * 8 + r) * D_ + et + r16] = acc[r] * SCALE_;
}

// ---------------------------------------------------------------------------
// Reduce NSPLIT kv partials -> kv[b,h,d,e]
// ---------------------------------------------------------------------------
__global__ __launch_bounds__(256) void kv_reduce(
    const float* __restrict__ part, float* __restrict__ out)
{
  const int i = blockIdx.x * 256 + threadIdx.x;   // 0 .. B*H*D*D-1
  float s = 0.f;
#pragma unroll
  for (int sp = 0; sp < NSPLIT_; ++sp)
    s += part[(size_t)sp * (B_ * H_ * D_ * D_) + i];
  out[i] = s;
}

// ---------------------------------------------------------------------------
// attn[b,s,h*64+e] = sum_d q[b,s,h*64+d] * kv[b,h,d,e]   (scale in kv)
// kv staged TRANSPOSED (kvT[e][d]) so the B-operand is aligned b128 loads.
// ---------------------------------------------------------------------------
__global__ __launch_bounds__(256) void attn_kernel(
    const unsigned short* __restrict__ Qb, const float* __restrict__ KV,
    unsigned short* __restrict__ Ob)
{
  __shared__ unsigned short q_lds[128][72];   // [m][d], 144B rows (16B aligned)
  __shared__ unsigned short kvT[64][72];      // [e][d]

  const int tid  = threadIdx.x;
  const int lane = tid & 31;
  const int wave = tid >> 5;              // 0..7 -> M tile
  const int mrow = blockIdx.x * 128;      // global row in [0, B*S)
  const int h    = blockIdx.y;
  const int b    = mrow / S_;
  const int lh = lane >> 4, r16 = lane & 15;

  { // stage q tile 128x64 (b128 copies)
    const int row = tid >> 1, seg = (tid & 1) << 5;
    const unsigned short* src = Qb + (size_t)(mrow + row) * E_ + h * D_ + seg;
#pragma unroll
    for (int j = 0; j < 4; ++j)
      *(uint4*)&q_lds[row][seg + j * 8] = *(const uint4*)(src + j * 8);
  }
  { // stage kv 64x64 f32 -> bf16, transposed
    const int d = tid >> 2, seg = (tid & 3) << 4;
    const float* src = KV + (size_t)(b * H_ + h) * D_ * D_ + d * D_ + seg;
#pragma unroll
    for (int j = 0; j < 16; ++j) kvT[seg + j][d] = f2bf(src[j]);
  }
  __syncthreads();

  const int m = wave * 16;
  Frag af[2];
#pragma unroll
  for (int c = 0; c < 2; ++c) {
    const unsigned short* r = q_lds[m + r16] + c * 32;
    af[c].q[0] = *(const uint4*)(r + 8 * lh);
    af[c].q[1] = *(const uint4*)(r + 16 + 8 * lh);
  }

#pragma unroll
  for (int nt = 0; nt < 4; ++nt) {
    v8f acc = {};
#pragma unroll
    for (int c = 0; c < 2; ++c) {
      Frag bfr;
      const unsigned short* r = kvT[nt * 16 + r16] + c * 32;
      bfr.q[0] = *(const uint4*)(r + 8 * lh);
      bfr.q[1] = *(const uint4*)(r + 16 + 8 * lh);
      acc = __builtin_amdgcn_wmma_f32_16x16x32_bf16(
          false, af[c].v, false, bfr.v, (short)0, acc, false, false);
    }
#pragma unroll
    for (int r = 0; r < 8; ++r) {
      const int mm = mrow + m + lh * 8 + r;
      Ob[(size_t)mm * E_ + h * D_ + nt * 16 + r16] = f2bf(acc[r]);
    }
  }
}

// ---------------------------------------------------------------------------
extern "C" void kernel_launch(void* const* d_in, const int* in_sizes, int n_in,
                              void* d_out, int out_size, void* d_ws, size_t ws_size,
                              hipStream_t stream)
{
  (void)in_sizes; (void)n_in; (void)out_size; (void)ws_size;
  const float* x   = (const float*)d_in[0];
  const float* q_w = (const float*)d_in[1];
  const float* q_b = (const float*)d_in[2];
  const float* k_w = (const float*)d_in[3];
  const float* k_b = (const float*)d_in[4];
  const float* v_w = (const float*)d_in[5];
  const float* v_b = (const float*)d_in[6];
  const float* o_w = (const float*)d_in[7];
  const float* o_b = (const float*)d_in[8];

  const size_t nX = (size_t)B_ * S_ * E_;   // 16,777,216
  const size_t nW = (size_t)E_ * E_;        // 1,048,576
  unsigned short* xb  = (unsigned short*)d_ws;       // bf16 x
  unsigned short* wq  = xb + nX;
  unsigned short* wk  = wq + nW;
  unsigned short* wv  = wk + nW;
  unsigned short* wo  = wv + nW;
  unsigned short* qb16 = wo + nW;                    // bf16 Q
  unsigned short* kb16 = qb16 + nX;                  // bf16 K
  unsigned short* vb16 = kb16 + nX;                  // bf16 V
  float* kvpart = (float*)(vb16 + nX);               // [NSPLIT,B,H,64,64]
  float* kvbuf  = kvpart + (size_t)NSPLIT_ * B_ * H_ * D_ * D_;
  unsigned short* attn = kb16;                       // reuse dead K buffer

  // one-time bf16 conversions
  cvt_f32_bf16<<<(int)(nX / 4 / 256), 256, 0, stream>>>(x,   xb, (int)(nX / 4));
  cvt_f32_bf16<<<(int)(nW / 4 / 256), 256, 0, stream>>>(q_w, wq, (int)(nW / 4));
  cvt_f32_bf16<<<(int)(nW / 4 / 256), 256, 0, stream>>>(k_w, wk, (int)(nW / 4));
  cvt_f32_bf16<<<(int)(nW / 4 / 256), 256, 0, stream>>>(v_w, wv, (int)(nW / 4));
  cvt_f32_bf16<<<(int)(nW / 4 / 256), 256, 0, stream>>>(o_w, wo, (int)(nW / 4));

  const int M = B_ * S_, N = E_, K = E_;
  dim3 gg(N / 128, M / 128);

  gemm_tn<true><<<gg, 512, 0, stream>>>(xb, wq, q_b, qb16, M, N, K);
  gemm_tn<true><<<gg, 512, 0, stream>>>(xb, wk, k_b, kb16, M, N, K);
  gemm_tn<true><<<gg, 512, 0, stream>>>(xb, wv, v_b, vb16, M, N, K);
  kv_kernel<<<dim3(B_ * H_, NSPLIT_), 512, 0, stream>>>(kb16, vb16, kvpart);
  kv_reduce<<<(B_ * H_ * D_ * D_) / 256, 256, 0, stream>>>(kvpart, kvbuf);
  attn_kernel<<<dim3(M / 128, H_), 256, 0, stream>>>(qb16, kvbuf, attn);
  gemm_tn<false><<<gg, 512, 0, stream>>>(attn, wo, o_b, d_out, M, N, K);
}
